// LSTMModelTimeShap_79053168050589
// MI455X (gfx1250) — matrix operand order repaired and checked by reading, not compiled, not run on hardware
//
#include <hip/hip_runtime.h>
#include <hip/hip_bf16.h>
#include <stdint.h>

// Problem dims (match reference)
#define B_  256
#define T_  512
#define I_  64
#define H_  512
#define HBUF (B_*H_)     // 131072 elems per parity buffer
#define XROW (T_*I_)     // 32768 elems per batch row of x
#define NWG 64
#define NTHREADS 256

typedef __attribute__((ext_vector_type(16))) __bf16 v16bf;
typedef __attribute__((ext_vector_type(8)))  float  v8f;
typedef __attribute__((ext_vector_type(4)))  float  f32x4;

union V16 { v16bf bf; f32x4 q[2]; };

__device__ __forceinline__ unsigned short f2bf(float f) {
  unsigned u = __builtin_bit_cast(unsigned, f);
  u += 0x7FFFu + ((u >> 16) & 1u);          // round-to-nearest-even
  return (unsigned short)(u >> 16);
}
__device__ __forceinline__ float bf2f(unsigned short h) {
  return __builtin_bit_cast(float, ((unsigned)h) << 16);
}
__device__ __forceinline__ float sigmoidf_(float x) { return 1.0f / (1.0f + __expf(-x)); }
__device__ __forceinline__ float tanhf_(float x)    { return 2.0f * sigmoidf_(2.0f * x) - 1.0f; }

// Device-wide barrier: monotonic counter (reset by init kernel each launch).
__device__ __forceinline__ void grid_barrier(unsigned* cnt, unsigned expect) {
  __syncthreads();
  if (threadIdx.x == 0) {
    atomicAdd(cnt, 1u);
    while (atomicAdd(cnt, 0u) < expect) __builtin_amdgcn_s_sleep(2);
  }
  __syncthreads();
}

// ---------------- preprocessing kernels ----------------

__global__ void init_state(unsigned short* h0, unsigned short* h1, unsigned* bar) {
  int idx = blockIdx.x * blockDim.x + threadIdx.x;
  if (idx < HBUF) {            // zero the parity-1 (t = -1) buffers
    h0[HBUF + idx] = 0;
    h1[HBUF + idx] = 0;
  }
  if (idx == 0) *bar = 0u;
}

__global__ void cvt_x(const float* __restrict__ x, unsigned short* __restrict__ xbf) {
  int idx = blockIdx.x * blockDim.x + threadIdx.x;
  xbf[idx] = f2bf(x[idx]);
}

// Pack W (rows = 4H output cols, ld = K-src width) into WMMA-B tile layout:
// tile (nt in [0,128), kc) is 32 lanes x 16 bf16 contiguous (32B/lane):
//   lane L<16 : col n = nt*16+L,      K = kc*32 + 0..15
//   lane L>=16: col n = nt*16+(L-16), K = kc*32 + 16..31
__global__ void pack_w(const float* __restrict__ W, unsigned short* __restrict__ PB,
                       int KC, int kc0, int nkc, int ld) {
  int j = blockIdx.x * blockDim.x + threadIdx.x;
  int total = 128 * nkc * 32;
  if (j >= total) return;
  int lane = j & 31;
  int kcl  = (j >> 5) % nkc;
  int nt   = (j >> 5) / nkc;
  int n  = nt * 16 + (lane & 15);
  int kb = kcl * 32 + (lane >> 4) * 16;
  unsigned short* dst = PB + ((size_t)nt * KC + (kc0 + kcl)) * 512 + lane * 16;
  const float* src = W + (size_t)n * ld + kb;
  #pragma unroll
  for (int q = 0; q < 16; ++q) dst[q] = f2bf(src[q]);
}

// ---------------- fused per-step gate GEMM + LSTM cell ----------------
// KC total K-chunks (32 each, KC must be even); kc < KC0 reads p0 (stride ld0),
// else p1 (stride H_).
template<int KC, int KC0>
__device__ __forceinline__ void lstm_gates(
    const unsigned short* __restrict__ PB,
    const unsigned short* __restrict__ p0, size_t ld0,
    const unsigned short* __restrict__ p1,
    unsigned short* __restrict__ lds,          // 2 x 4 x 512 ushorts (8KB)
    int nt, int lane, int khalf, int m0, int ncol,
    const float* bias, float* c,
    unsigned short* __restrict__ hout)
{
  const int tid  = threadIdx.x;
  const int arow = lane & 15;

  // async-stage one 4KB B chunk (4 gate tiles) into LDS buffer (kc&1):
  // 256 threads x one global_load_async_to_lds_b128 each (ASYNCcnt-tracked,
  // bypasses VGPRs entirely).
  auto stageB = [&](int kc) {
    int g = tid >> 6, j = tid & 63;
    const unsigned short* src =
        PB + ((size_t)(g * 32 + nt) * KC + kc) * 512 + j * 8;
    unsigned ldsoff =
        (unsigned)(uintptr_t)(lds + ((kc & 1) * 2048) + g * 512 + j * 8);
    unsigned long long ga = (unsigned long long)(uintptr_t)src;
    asm volatile("global_load_async_to_lds_b128 %0, %1, off"
                 :: "v"(ldsoff), "v"(ga) : "memory");
  };
  // A tile 16x32 bf16: lane L row M=L%16; khalf=0 -> K{0..15}, khalf=1 -> K{16..31}
  auto loadA = [&](int kc) -> V16 {
    const unsigned short* ap = (kc < KC0)
        ? p0 + (size_t)(m0 + arow) * ld0 + kc * 32
        : p1 + (size_t)(m0 + arow) * H_ + (kc - KC0) * 32;
    V16 r;
    r.q[0] = *(const f32x4*)(ap + khalf * 16);
    r.q[1] = *(const f32x4*)(ap + khalf * 16 + 8);
    return r;
  };

  v8f acc[4];
  #pragma unroll
  for (int g = 0; g < 4; ++g) {
    #pragma unroll
    for (int e = 0; e < 8; ++e) acc[g][e] = bias[g];
  }

  // one pipelined K-chunk: consume acur, prefetch chunk kc+1 into anext
  auto chunk = [&](int kc, V16& acur, V16& anext) {
    asm volatile("s_wait_asynccnt 0x0" ::: "memory");  // staged chunk landed
    __syncthreads();                                   // ... and published WG-wide
    if (kc + 1 < KC) stageB(kc + 1);                   // prefetch next B chunk
    if (kc + 1 < KC) anext = loadA(kc + 1);            // prefetch next A tile
    // all 4 gate B tiles into distinct regs -> 8 ds_load_b128 in flight,
    // then 4 back-to-back WMMAs (disjoint accumulators, no D->A/B hazard)
    const unsigned short* lb = lds + (kc & 1) * 2048 + lane * 16;
    V16 b0, b1, b2, b3;
    b0.q[0] = ((const f32x4*)(lb        ))[0]; b0.q[1] = ((const f32x4*)(lb        ))[1];
    b1.q[0] = ((const f32x4*)(lb +  512))[0]; b1.q[1] = ((const f32x4*)(lb +  512))[1];
    b2.q[0] = ((const f32x4*)(lb + 1024))[0]; b2.q[1] = ((const f32x4*)(lb + 1024))[1];
    b3.q[0] = ((const f32x4*)(lb + 1536))[0]; b3.q[1] = ((const f32x4*)(lb + 1536))[1];
    acc[0] = __builtin_amdgcn_wmma_f32_16x16x32_bf16(false, acur.bf, false, b0.bf, (short)0, acc[0], false, false);
    acc[1] = __builtin_amdgcn_wmma_f32_16x16x32_bf16(false, acur.bf, false, b1.bf, (short)0, acc[1], false, false);
    acc[2] = __builtin_amdgcn_wmma_f32_16x16x32_bf16(false, acur.bf, false, b2.bf, (short)0, acc[2], false, false);
    acc[3] = __builtin_amdgcn_wmma_f32_16x16x32_bf16(false, acur.bf, false, b3.bf, (short)0, acc[3], false, false);
  };

  stageB(0);
  V16 a0 = loadA(0), a1;

  // 2x-unrolled ping-pong pipeline: no A-register copies, no WMMA->VALU
  // hazard NOPs from shuffling the operand registers.
  #pragma unroll 1
  for (int kc = 0; kc < KC; kc += 2) {
    chunk(kc,     a0, a1);
    chunk(kc + 1, a1, a0);
  }
  __syncthreads();   // all waves done reading LDS before caller restages
  // LSTM cell update; C/D layout: elem e -> (M = e + khalf*8, N = lane%16)
  #pragma unroll
  for (int e = 0; e < 8; ++e) {
    float ig = sigmoidf_(acc[0][e]);
    float fg = sigmoidf_(acc[1][e]);
    float gg = tanhf_(acc[2][e]);
    float og = sigmoidf_(acc[3][e]);
    float cn = fg * c[e] + ig * gg;
    c[e] = cn;
    float hv = og * tanhf_(cn);
    hout[(size_t)(m0 + e + khalf * 8) * H_ + ncol] = f2bf(hv);
  }
}

// ---------------- persistent kernel: all 512 steps, both layers, FC ----------------
__global__ void __launch_bounds__(NTHREADS, 1)
lstm_persistent(const unsigned short* __restrict__ xbf,
                const unsigned short* __restrict__ PB0,
                const unsigned short* __restrict__ PB1,
                const float* __restrict__ b_ih0, const float* __restrict__ b_hh0,
                const float* __restrict__ b_ih1, const float* __restrict__ b_hh1,
                const float* __restrict__ Wfc,  const float* __restrict__ bfc,
                unsigned short* __restrict__ h0, unsigned short* __restrict__ h1,
                unsigned* __restrict__ bar, float* __restrict__ out)
{
  __shared__ unsigned short lds[2 * 4 * 512];   // 8KB double-buffered B stage
  const int tid  = threadIdx.x;
  const int lane = tid & 31, wave = tid >> 5;
  const int wg   = blockIdx.x;
  const int nt   = wg & 31, mg = wg >> 5;       // 32 n-tiles x 2 m-groups
  const int m0   = (mg * 8 + wave) * 16;        // 16 batch rows per wave
  const int khalf = lane >> 4;
  const int ncol  = nt * 16 + (lane & 15);

  float bias0[4], bias1[4];
  #pragma unroll
  for (int g = 0; g < 4; ++g) {
    bias0[g] = b_ih0[g * H_ + ncol] + b_hh0[g * H_ + ncol];
    bias1[g] = b_ih1[g * H_ + ncol] + b_hh1[g * H_ + ncol];
  }
  float c0[8], c1[8];
  #pragma unroll
  for (int e = 0; e < 8; ++e) { c0[e] = 0.f; c1[e] = 0.f; }

  unsigned bseq = 0;
  #pragma unroll 1
  for (int t = 0; t < T_; ++t) {
    const int wp = t & 1, rp = wp ^ 1;
    // layer 0: gates = x_t @ Wih0^T + h0_prev @ Whh0^T + b   (K = 64 + 512)
    lstm_gates<18, 2>(PB0, xbf + (size_t)t * I_, (size_t)XROW, h0 + rp * HBUF,
                      lds, nt, lane, khalf, m0, ncol, bias0, c0, h0 + wp * HBUF);
    __builtin_amdgcn_fence(__ATOMIC_RELEASE, "agent");
    grid_barrier(bar, (++bseq) * gridDim.x);    // h0_t visible device-wide
    __builtin_amdgcn_fence(__ATOMIC_ACQUIRE, "agent");
    // layer 1: gates = h0_t @ Wih1^T + h1_prev @ Whh1^T + b  (K = 512 + 512)
    lstm_gates<32, 16>(PB1, h0 + wp * HBUF, (size_t)H_, h1 + rp * HBUF,
                       lds, nt, lane, khalf, m0, ncol, bias1, c1, h1 + wp * HBUF);
  }
  __builtin_amdgcn_fence(__ATOMIC_RELEASE, "agent");
  grid_barrier(bar, (++bseq) * gridDim.x);
  __builtin_amdgcn_fence(__ATOMIC_ACQUIRE, "agent");

  if (wg == 0) {  // FC epilogue: out[b] = h1_last[b,:] . Wfc + bfc
    const unsigned short* hr = h1 + HBUF + (size_t)tid * H_;   // t=511 -> parity 1
    float s = bfc[0];
    #pragma unroll 4
    for (int k = 0; k < H_; ++k) s += bf2f(hr[k]) * Wfc[k];
    out[tid] = s;
  }
}

// ---------------- host launcher ----------------
extern "C" void kernel_launch(void* const* d_in, const int* in_sizes, int n_in,
                              void* d_out, int out_size, void* d_ws, size_t ws_size,
                              hipStream_t stream) {
  (void)in_sizes; (void)n_in; (void)out_size; (void)ws_size;
  const float* x     = (const float*)d_in[0];
  const float* W_ih0 = (const float*)d_in[1];
  const float* W_hh0 = (const float*)d_in[2];
  const float* b_ih0 = (const float*)d_in[3];
  const float* b_hh0 = (const float*)d_in[4];
  const float* W_ih1 = (const float*)d_in[5];
  const float* W_hh1 = (const float*)d_in[6];
  const float* b_ih1 = (const float*)d_in[7];
  const float* b_hh1 = (const float*)d_in[8];
  const float* W_fc  = (const float*)d_in[9];
  const float* b_fc  = (const float*)d_in[10];
  float* out = (float*)d_out;

  // Workspace layout (~24.4 MB total)
  char* ws = (char*)d_ws;
  unsigned short* xbf = (unsigned short*)(ws);                  // 16,777,216 B
  unsigned short* PB0 = (unsigned short*)(ws + 16777216);       //  2,359,296 B (128 x 18 x 1KB)
  unsigned short* PB1 = (unsigned short*)(ws + 19136512);       //  4,194,304 B (128 x 32 x 1KB)
  unsigned short* h0  = (unsigned short*)(ws + 23330816);       //    524,288 B (2 parities)
  unsigned short* h1  = (unsigned short*)(ws + 23855104);       //    524,288 B
  unsigned*       bar = (unsigned*)      (ws + 24379392);       //         64 B

  init_state<<<dim3((HBUF + 255) / 256), dim3(256), 0, stream>>>(h0, h1, bar);
  cvt_x<<<dim3(B_ * T_ * I_ / 256), dim3(256), 0, stream>>>(x, xbf);
  pack_w<<<dim3((128 *  2 * 32 + 255) / 256), dim3(256), 0, stream>>>(W_ih0, PB0, 18,  0,  2,  64);
  pack_w<<<dim3((128 * 16 * 32 + 255) / 256), dim3(256), 0, stream>>>(W_hh0, PB0, 18,  2, 16, 512);
  pack_w<<<dim3((128 * 16 * 32 + 255) / 256), dim3(256), 0, stream>>>(W_ih1, PB1, 32,  0, 16, 512);
  pack_w<<<dim3((128 * 16 * 32 + 255) / 256), dim3(256), 0, stream>>>(W_hh1, PB1, 32, 16, 16, 512);

  lstm_persistent<<<dim3(NWG), dim3(NTHREADS), 0, stream>>>(
      xbf, PB0, PB1, b_ih0, b_hh0, b_ih1, b_hh1, W_fc, b_fc, h0, h1, bar, out);
}
